// spaceGraph_72353019069085
// MI455X (gfx1250) — matrix operand order repaired and checked by reading, not compiled, not run on hardware
//
#include <hip/hip_runtime.h>

typedef _Float16 f16;
typedef __attribute__((ext_vector_type(16))) _Float16 v16h;
typedef __attribute__((ext_vector_type(8)))  _Float16 v8h;
typedef __attribute__((ext_vector_type(8)))  float    v8f;

#define NHEADS 8
#define NVARS  128
#define HEADD  64
#define PATCH  64       // P = L / N
#define DMODEL 512      // H * HEAD
#define LSEQ   8192

// ---- LDS layout (bytes). Region [0,37376) is dual-use:
//   phase 1-2:  xh[128][72] f16 (18432) | w1h[64][72] (9216) | w2h[64][72] (9216) | b1 (256) | b2 (256)
//   phase 3-4:  S[128][136] f16 (34816)   (xh/w1h/w2h dead after the projections)
#define LDS_XH    0
#define LDS_W1    18432
#define LDS_W2    27648
#define LDS_B1    36864
#define LDS_B2    37120
#define LDS_S     0
#define LDS_XHT   37376   // xh^T [64][136] f16 = 17408
#define LDS_Q     54784   // q [128][72] f16 = 18432
#define LDS_K     73216   // k [128][72] f16 = 18432
#define LDS_TOTAL 91648

// A-fragment (16x32, f16): lane = m + 16*hs holds A[m][k0+8*hs .. +8) in v[0..7]
// and A[m][k0+16+8*hs .. +8) in v[8..15]  (per ISA 16-bit A layout).
__device__ __forceinline__ v16h load_a(const f16* __restrict__ base, int ld,
                                       int row0, int k0, int lane) {
  const int m = lane & 15, hs = lane >> 4;
  const f16* p = base + (row0 + m) * ld + k0 + 8 * hs;
  v8h lo = *(const v8h*)(p);
  v8h hi = *(const v8h*)(p + 16);
  v16h a;
#pragma unroll
  for (int i = 0; i < 8; ++i) { a[i] = lo[i]; a[i + 8] = hi[i]; }
  return a;
}

// B-fragment (32x16, f16) from a [n][k]-major buffer: lane = n + 16*kh holds
// B[k0+8*kh .. +8)[n0+n] in v[0..7] and B[k0+16+8*kh .. +8)[n0+n] in v[8..15].
__device__ __forceinline__ v16h load_b(const f16* __restrict__ base, int ld,
                                       int n0, int k0, int lane) {
  const int n = lane & 15, kh = lane >> 4;
  const f16* p = base + (n0 + n) * ld + k0 + 8 * kh;
  v8h lo = *(const v8h*)(p);
  v8h hi = *(const v8h*)(p + 16);
  v16h b;
#pragma unroll
  for (int i = 0; i < 8; ++i) { b[i] = lo[i]; b[i + 8] = hi[i]; }
  return b;
}

__device__ __forceinline__ v8f wmma16(v16h a, v16h b, v8f c) {
  return __builtin_amdgcn_wmma_f32_16x16x32_f16(false, a, false, b, (short)0, c,
                                                false, false);
}

__global__ __launch_bounds__(256)
void spacegraph_kernel(const float* __restrict__ x,
                       const float* __restrict__ w1, const float* __restrict__ b1,
                       const float* __restrict__ w2, const float* __restrict__ b2,
                       float* __restrict__ out) {
  __shared__ __align__(16) unsigned char smem[LDS_TOTAL];
  f16*   xh  = (f16*)(smem + LDS_XH);
  f16*   w1h = (f16*)(smem + LDS_W1);
  f16*   w2h = (f16*)(smem + LDS_W2);
  float* b1s = (float*)(smem + LDS_B1);
  float* b2s = (float*)(smem + LDS_B2);
  f16*   Sh  = (f16*)(smem + LDS_S);
  f16*   xhT = (f16*)(smem + LDS_XHT);
  f16*   qh  = (f16*)(smem + LDS_Q);
  f16*   kh  = (f16*)(smem + LDS_K);

  const int tid  = threadIdx.x;
  const int lane = tid & 31;
  const int wave = tid >> 5;
  const int ln   = lane & 15;
  const int hs   = lane >> 4;

  const int blk = blockIdx.x;           // b*512 + h*64 + p
  const int b = blk >> 9;
  const int h = (blk >> 6) & 7;
  const int p = blk & 63;

  const float* xbase = x + ((size_t)b * LSEQ + p) * DMODEL + h * HEADD;

  // ---------- Phase 1: global -> LDS (f32 -> f16), build xh and xh^T ----------
#pragma unroll
  for (int e = 0; e < 8; ++e) {
    int e4 = tid + e * 256;             // 0..2047 float4s of the 128x64 tile
    int i  = e4 >> 4;                   // variable row
    int d4 = (e4 & 15) << 2;            // head-dim
    const float4 v = *(const float4*)(xbase + (size_t)i * (PATCH * DMODEL) + d4);
    f16 h0 = (f16)v.x, h1 = (f16)v.y, h2 = (f16)v.z, h3 = (f16)v.w;
    f16* dst = xh + i * 72 + d4;
    dst[0] = h0; dst[1] = h1; dst[2] = h2; dst[3] = h3;
    xhT[(d4 + 0) * 136 + i] = h0;
    xhT[(d4 + 1) * 136 + i] = h1;
    xhT[(d4 + 2) * 136 + i] = h2;
    xhT[(d4 + 3) * 136 + i] = h3;
  }
#pragma unroll
  for (int e = 0; e < 4; ++e) {
    int e4 = tid + e * 256;             // 0..1023 float4s of each 64x64 weight
    int r  = e4 >> 4;
    int c4 = (e4 & 15) << 2;
    float4 a = *(const float4*)(w1 + r * 64 + c4);
    float4 c = *(const float4*)(w2 + r * 64 + c4);
    f16* d1 = w1h + r * 72 + c4;
    f16* d2 = w2h + r * 72 + c4;
    d1[0] = (f16)a.x; d1[1] = (f16)a.y; d1[2] = (f16)a.z; d1[3] = (f16)a.w;
    d2[0] = (f16)c.x; d2[1] = (f16)c.y; d2[2] = (f16)c.z; d2[3] = (f16)c.w;
  }
  if (tid < 64)            b1s[tid]      = b1[tid];
  else if (tid < 128)      b2s[tid - 64] = b2[tid - 64];
  __syncthreads();

  // ---------- Phase 2: q = xh*w1^T + b1 ; k = xh*w2^T + b2 (wave owns 16 rows) ----------
  {
    const int row0 = wave * 16;
#pragma unroll
    for (int nt = 0; nt < 4; ++nt) {
      const int n0 = nt * 16;
      v8f cq = {}; v8f ck = {};
#pragma unroll
      for (int kt = 0; kt < 2; ++kt) {
        const int k0 = kt * 32;
        v16h a  = load_a(xh, 72, row0, k0, lane);
        v16h bq = load_b(w1h, 72, n0, k0, lane);   // w1 row-major == [n][k] for x*w^T
        v16h bk = load_b(w2h, 72, n0, k0, lane);
        cq = wmma16(a, bq, cq);
        ck = wmma16(a, bk, ck);
      }
      const float biasq = b1s[n0 + ln];
      const float biask = b2s[n0 + ln];
#pragma unroll
      for (int r = 0; r < 8; ++r) {
        const int row = row0 + r + 8 * hs;         // C layout: VGPR r -> row r / r+8
        qh[row * 72 + n0 + ln] = (f16)(cq[r] + biasq);
        kh[row * 72 + n0 + ln] = (f16)(ck[r] + biask);
      }
    }
  }
  __syncthreads();   // also fences the xh/w1h -> Sh alias reuse

  // ---------- Phase 3: S = gelu(q*k^T), softmax rows (L1 renorm after softmax == identity) ----------
  {
    const int row0 = wave * 16;
    v8f acc[8];
#pragma unroll
    for (int jt = 0; jt < 8; ++jt) {
      v8f c = {};
#pragma unroll
      for (int kt = 0; kt < 2; ++kt) {
        v16h a  = load_a(qh, 72, row0, kt * 32, lane);
        v16h bf = load_b(kh, 72, jt * 16, kt * 32, lane);  // k stored [j][d] == [n][k]
        c = wmma16(a, bf, c);
      }
#pragma unroll
      for (int r = 0; r < 8; ++r) {                 // exact GELU: 0.5x(1+erf(x/sqrt2))
        float v = c[r];
        c[r] = 0.5f * v * (1.0f + erff(v * 0.70710678118654752f));
      }
      acc[jt] = c;
    }
    // Row r lives in one 16-lane half (lanes 0-15 -> row0+r, lanes 16-31 -> row0+r+8):
    // reduce across 8 j-tiles in registers, then across 16 lanes with shfl_xor.
#pragma unroll
    for (int r = 0; r < 8; ++r) {
      float m = acc[0][r];
#pragma unroll
      for (int jt = 1; jt < 8; ++jt) m = fmaxf(m, acc[jt][r]);
      m = fmaxf(m, __shfl_xor(m, 8, 32));
      m = fmaxf(m, __shfl_xor(m, 4, 32));
      m = fmaxf(m, __shfl_xor(m, 2, 32));
      m = fmaxf(m, __shfl_xor(m, 1, 32));
      float s = 0.f;
#pragma unroll
      for (int jt = 0; jt < 8; ++jt) {
        float e = __expf(acc[jt][r] - m);
        acc[jt][r] = e;
        s += e;
      }
      s += __shfl_xor(s, 8, 32);
      s += __shfl_xor(s, 4, 32);
      s += __shfl_xor(s, 2, 32);
      s += __shfl_xor(s, 1, 32);
      const float inv = 1.0f / s;
      const int row = row0 + r + 8 * hs;
#pragma unroll
      for (int jt = 0; jt < 8; ++jt)
        Sh[row * 136 + jt * 16 + ln] = (f16)(acc[jt][r] * inv);
    }
  }
  __syncthreads();

  // ---------- Phase 4: out = S * xh  (K = 128 over j; B operand from xh^T) ----------
  {
    const int row0 = wave * 16;
    float* obase = out + ((size_t)b * LSEQ + p) * DMODEL + h * HEADD;
#pragma unroll
    for (int nt = 0; nt < 4; ++nt) {
      v8f c = {};
#pragma unroll
      for (int kt = 0; kt < 4; ++kt) {
        v16h a  = load_a(Sh, 136, row0, kt * 32, lane);
        v16h bf = load_b(xhT, 136, nt * 16, kt * 32, lane);  // xh^T is [d][j] == [n][k]
        c = wmma16(a, bf, c);
      }
      const int dcol = nt * 16 + ln;
#pragma unroll
      for (int r = 0; r < 8; ++r) {
        const int i = row0 + r + 8 * hs;
        obase[(size_t)i * (PATCH * DMODEL) + dcol] = c[r];
      }
    }
  }
}

extern "C" void kernel_launch(void* const* d_in, const int* in_sizes, int n_in,
                              void* d_out, int out_size, void* d_ws, size_t ws_size,
                              hipStream_t stream) {
  const float* x  = (const float*)d_in[0];
  const float* w1 = (const float*)d_in[1];
  const float* b1 = (const float*)d_in[2];
  const float* w2 = (const float*)d_in[3];
  const float* b2 = (const float*)d_in[4];
  float* out = (float*)d_out;

  const int Bsz = in_sizes[0] / (LSEQ * DMODEL);   // = 4
  dim3 grid(Bsz * NHEADS * PATCH);                 // one block per (b, h, patch)
  spacegraph_kernel<<<grid, 256, 0, stream>>>(x, w1, b1, w2, b2, out);
}
